// CNF_42047729827878
// MI455X (gfx1250) — matrix operand order repaired and checked by reading, not compiled, not run on hardware
//
#include <hip/hip_runtime.h>

typedef __attribute__((ext_vector_type(16))) _Float16 v16h;
typedef __attribute__((ext_vector_type(8)))  _Float16 v8h;
typedef __attribute__((ext_vector_type(8)))  float    v8f;

#define BN    524288
#define HID   64
#define NINT  9            // STEPS-1 intervals

// ---------------- helpers ----------------

__device__ __forceinline__ float ftanh(float x) {
#if __has_builtin(__builtin_amdgcn_tanhf)
  return __builtin_amdgcn_tanhf(x);        // v_tanh_f32 on gfx1250
#else
  float xc = fminf(fmaxf(x, -9.0f), 9.0f);
  float e  = __builtin_amdgcn_exp2f(xc * 2.8853900817779268f);   // exp(2x)
  return (e - 1.0f) * __builtin_amdgcn_rcpf(e + 1.0f);
#endif
}

// within-wave LDS producer->consumer fence (ds ops are in-order per wave;
// this stops compiler reordering and drains DScnt)
__device__ __forceinline__ void wave_lds_fence() {
  __builtin_amdgcn_wave_barrier();
  asm volatile("s_wait_dscnt 0x0" ::: "memory");
  __builtin_amdgcn_wave_barrier();
}

// async copy: one f32 per lane, global -> LDS (ASYNCcnt path)
__device__ __forceinline__ void async_f32_to_lds(unsigned lds_off, const float* gaddr) {
  asm volatile("global_load_async_to_lds_b32 %0, %1, off"
               :: "v"(lds_off), "v"(gaddr) : "memory");
}
__device__ __forceinline__ void wait_async0() {
  asm volatile("s_wait_asynccnt 0x0" ::: "memory");
}

__device__ __forceinline__ v16h join16(v8h a, v8h b) {
  return __builtin_shufflevector(a, b, 0,1,2,3,4,5,6,7,8,9,10,11,12,13,14,15);
}

// B fragment: packed per-lane contiguous 32B in LDS -> 2x ds_load_b128
__device__ __forceinline__ v16h load_frag(const _Float16* p) {
  return *(const v16h*)p;
}

// A fragment from row-major 16x64 f16 staging buffer (ISA 16-bit A layout):
// lane L: row M = L%16, khalf = (L/16)*8; j=0..7 -> K=kt*32+khalf+j,
// j=8..15 -> K=kt*32+16+khalf+(j-8)
__device__ __forceinline__ v16h load_Astage(const _Float16* st, int lane, int kt) {
  const int m     = lane & 15;
  const int khalf = (lane >> 4) * 8;
  const _Float16* p = st + m * 64 + kt * 32 + khalf;
  v8h a = *(const v8h*)(p);
  v8h b = *(const v8h*)(p + 16);
  return join16(a, b);
}

#define WMMA_F16(A, B, C) \
  __builtin_amdgcn_wmma_f32_16x16x32_f16(false, (A), false, (B), (short)0, (C), false, false)

// ---------------- kernel ----------------

__global__ __launch_bounds__(256) void cnf_rk4_wmma(
    const float* __restrict__ x,   const float* __restrict__ ctx,
    const float* __restrict__ eps, const float* __restrict__ W1,
    const float* __restrict__ b1,  const float* __restrict__ W2,
    const float* __restrict__ b2,  const float* __restrict__ W3,
    const float* __restrict__ b3,  float* __restrict__ out)
{
  // 24 B-layout fragments (8 fwd W2, 8 bwd W2^T, 8 ctx W1) : 24 KB
  __shared__ __align__(32) _Float16 fragLDS[24 * 512];
  // per-wave 16x64 f16 staging for C-layout -> A-layout transpose : 16 KB
  __shared__ __align__(32) _Float16 stageLDS[8][16 * 64];
  // per-wave double-buffered eps staging (async-filled) : 2 KB
  __shared__ __align__(16) float epsLDS[8][2][32];

  const int lane  = threadIdx.x & 31;
  const int wv    = threadIdx.x >> 5;
  const int lo    = lane & 15;
  const int hi    = lane >> 4;
  const int khalf = hi * 8;     // A-layout K sub-block
  const int mbase = hi * 8;     // C-layout row group
  const int s0    = (blockIdx.x * 8 + wv) * 16;   // first sample of this wave's tile

  // ---- pack W2 / W2^T / W1[ctx] into WMMA B-layout fragments in LDS ----
  // B layout: lane holds column n = lane%16, K = kt*32 + (j<8 ? (lane/16)*8 + j : 16 + (lane/16)*8 + j-8)
  for (int f = wv; f < 24; f += 8) {
    const int kind = f >> 3;            // 0: W2 fwd, 1: W2^T bwd, 2: W1 ctx
    const int r  = f & 7;
    const int nt = r >> 1, kt = r & 1;
    const int n  = nt * 16 + lo;
    _Float16* dst = &fragLDS[f * 512 + lane * 16];
#pragma unroll
    for (int j = 0; j < 16; ++j) {
      const int k = kt * 32 + (j < 8 ? hi * 8 + j : 16 + hi * 8 + (j - 8));
      float v;
      if (kind == 0)      v = W2[k * HID + n];          // a2 = h1 @ W2
      else if (kind == 1) v = W2[n * HID + k];          // dh1 = da2 @ W2^T
      else                v = W1[(2 + k) * HID + n];    // cpre = ctx @ W1[2:66]
      dst[j] = (_Float16)v;
    }
  }
  __syncthreads();

  _Float16* st = stageLDS[wv];

  // ---- per-lane weight slices (column n = nt*16+lo) ----
  float w1a[4], w1b[4], w1t[4], b1v[4], b2v[4], w3a[4], w3b[4];
#pragma unroll
  for (int nt = 0; nt < 4; ++nt) {
    const int n = nt * 16 + lo;
    w1a[nt] = W1[0 * HID + n];
    w1b[nt] = W1[1 * HID + n];
    w1t[nt] = W1[66 * HID + n];
    b1v[nt] = b1[n];
    b2v[nt] = b2[n];
    w3a[nt] = W3[n * 2 + 0];
    w3b[nt] = W3[n * 2 + 1];
  }
  const float b30 = b3[0], b31 = b3[1];

  // ---- cpre = context @ W1[2:66] + b1 (stage-invariant; 8 WMMA) ----
  v16h actx[2];
#pragma unroll
  for (int kt = 0; kt < 2; ++kt) {
    const float* cp = ctx + (size_t)(s0 + lo) * 64 + kt * 32 + khalf;
    float4 f0 = *(const float4*)(cp + 0);
    float4 f1 = *(const float4*)(cp + 4);
    float4 f2 = *(const float4*)(cp + 16);
    float4 f3 = *(const float4*)(cp + 20);
    v16h a;
    a[0]=(_Float16)f0.x;  a[1]=(_Float16)f0.y;  a[2]=(_Float16)f0.z;  a[3]=(_Float16)f0.w;
    a[4]=(_Float16)f1.x;  a[5]=(_Float16)f1.y;  a[6]=(_Float16)f1.z;  a[7]=(_Float16)f1.w;
    a[8]=(_Float16)f2.x;  a[9]=(_Float16)f2.y;  a[10]=(_Float16)f2.z; a[11]=(_Float16)f2.w;
    a[12]=(_Float16)f3.x; a[13]=(_Float16)f3.y; a[14]=(_Float16)f3.z; a[15]=(_Float16)f3.w;
    actx[kt] = a;
  }
  v8f cpre[4];
#pragma unroll
  for (int nt = 0; nt < 4; ++nt) {
    v8f c = {};
    c = WMMA_F16(actx[0], load_frag(&fragLDS[(16 + nt * 2 + 0) * 512 + lane * 16]), c);
    c = WMMA_F16(actx[1], load_frag(&fragLDS[(16 + nt * 2 + 1) * 512 + lane * 16]), c);
#pragma unroll
    for (int r = 0; r < 8; ++r) c[r] += b1v[nt];
    cpre[nt] = c;
  }

  // ---- ODE state, fully in registers (replicated per 16-lane half) ----
  float zb0[8], zb1[8], lpacc[8];
  {
    const float* xp = x + (size_t)(s0 + mbase) * 2;
#pragma unroll
    for (int r = 0; r < 8; ++r) { zb0[r] = xp[2*r]; zb1[r] = xp[2*r+1]; lpacc[r] = 0.f; }
  }

  const float hstep = -1.0f / (float)NINT;

  for (int it = 0; it < NINT; ++it) {
    const float ti = 1.0f + hstep * (float)it;
    float az0[8], az1[8], al[8], ze0[8], ze1[8];
#pragma unroll
    for (int r = 0; r < 8; ++r) { az0[r]=0.f; az1[r]=0.f; al[r]=0.f; ze0[r]=zb0[r]; ze1[r]=zb1[r]; }

#pragma unroll
    for (int s = 0; s < 4; ++s) {
      const float ts = (s == 0) ? ti : (s == 3) ? (ti + hstep) : (ti + 0.5f * hstep);
      const int slot = s & 1;
      const float* ebuf = epsLDS[wv][slot];

      // kick off async eps copy for this stage: 32 consecutive floats / wave,
      // one b32 per lane, deduplicated through LDS (double-buffered on s&1)
      {
        const float* egp = eps + (((size_t)it * 4 + s) * (size_t)BN + (size_t)s0) * 2 + lane;
        const unsigned eoff = (unsigned)(uintptr_t)&epsLDS[wv][slot][lane];
        async_f32_to_lds(eoff, egp);
      }

      // layer 1: a1 = cpre + z*W1z + t*W1t ; h1 = tanh(a1)   (C layout)
      v8f h1[4];
#pragma unroll
      for (int nt = 0; nt < 4; ++nt) {
#pragma unroll
        for (int r = 0; r < 8; ++r) {
          float a = cpre[nt][r] + ze0[r] * w1a[nt] + ze1[r] * w1b[nt] + ts * w1t[nt];
          h1[nt][r] = ftanh(a);
        }
      }

      // C-layout f32 -> row-major f16 staging -> A-layout fragments
      wave_lds_fence();
#pragma unroll
      for (int nt = 0; nt < 4; ++nt) {
        const int n = nt * 16 + lo;
#pragma unroll
        for (int r = 0; r < 8; ++r) st[(mbase + r) * 64 + n] = (_Float16)h1[nt][r];
      }
      // h1 no longer needed as tanh value; keep (1 - h1^2) for the VJP
#pragma unroll
      for (int nt = 0; nt < 4; ++nt)
#pragma unroll
        for (int r = 0; r < 8; ++r) h1[nt][r] = 1.0f - h1[nt][r] * h1[nt][r];

      wave_lds_fence();
      v16h A0 = load_Astage(st, lane, 0);
      v16h A1 = load_Astage(st, lane, 1);

      // forward 64x64 GEMM: work = a2 = h1 @ W2   (8 WMMA)
      v8f work[4];
#pragma unroll
      for (int nt = 0; nt < 4; ++nt) {
        v8f c = {};
        c = WMMA_F16(A0, load_frag(&fragLDS[(nt * 2 + 0) * 512 + lane * 16]), c);
        c = WMMA_F16(A1, load_frag(&fragLDS[(nt * 2 + 1) * 512 + lane * 16]), c);
        work[nt] = c;
      }

      // eps is needed from here on
      wait_async0();

      // h2 = tanh(a2+b2); f-partials via W3; work <- d_a2 = (eps@W3^T)*(1-h2^2)
      float fp0[8], fp1[8];
#pragma unroll
      for (int r = 0; r < 8; ++r) {
        const float2 e = *(const float2*)&ebuf[2 * (mbase + r)];
        float f0 = 0.f, f1 = 0.f;
#pragma unroll
        for (int nt = 0; nt < 4; ++nt) {
          float h2 = ftanh(work[nt][r] + b2v[nt]);
          f0 = fmaf(h2, w3a[nt], f0);
          f1 = fmaf(h2, w3b[nt], f1);
          float dh2 = e.x * w3a[nt] + e.y * w3b[nt];
          work[nt][r] = dh2 * (1.0f - h2 * h2);
        }
        fp0[r] = f0; fp1[r] = f1;
      }

      // d_a2 through staging -> A layout
      wave_lds_fence();
#pragma unroll
      for (int nt = 0; nt < 4; ++nt) {
        const int n = nt * 16 + lo;
#pragma unroll
        for (int r = 0; r < 8; ++r) st[(mbase + r) * 64 + n] = (_Float16)work[nt][r];
      }
      wave_lds_fence();
      v16h D0 = load_Astage(st, lane, 0);
      v16h D1 = load_Astage(st, lane, 1);

      // backward 64x64 GEMM: work <- d_h1 = d_a2 @ W2^T   (8 WMMA)
#pragma unroll
      for (int nt = 0; nt < 4; ++nt) {
        v8f c = {};
        c = WMMA_F16(D0, load_frag(&fragLDS[(8 + nt * 2 + 0) * 512 + lane * 16]), c);
        c = WMMA_F16(D1, load_frag(&fragLDS[(8 + nt * 2 + 1) * 512 + lane * 16]), c);
        work[nt] = c;
      }

      // d_a1 = d_h1*(1-h1^2); div partial q = sum_n d_a1 * (W1z^T eps)
      float q[8];
#pragma unroll
      for (int r = 0; r < 8; ++r) {
        const float2 e = *(const float2*)&ebuf[2 * (mbase + r)];
        float qq = 0.f;
#pragma unroll
        for (int nt = 0; nt < 4; ++nt) {
          float da1 = work[nt][r] * h1[nt][r];     // h1 holds (1 - tanh^2)
          qq = fmaf(da1, w1a[nt] * e.x + w1b[nt] * e.y, qq);
        }
        q[r] = qq;
      }

      // butterfly reduce over each 16-lane half (N dimension)
#pragma unroll
      for (int mask = 1; mask <= 8; mask <<= 1) {
#pragma unroll
        for (int r = 0; r < 8; ++r) {
          fp0[r] += __shfl_xor(fp0[r], mask, 32);
          fp1[r] += __shfl_xor(fp1[r], mask, 32);
          q[r]   += __shfl_xor(q[r],   mask, 32);
        }
      }

      // RK4 accumulate + next evaluation point
      const float wrk = (s == 0 || s == 3) ? 1.0f : 2.0f;
      const float cnx = (s == 2) ? hstep : 0.5f * hstep;
#pragma unroll
      for (int r = 0; r < 8; ++r) {
        float kz0 = fp0[r] + b30;
        float kz1 = fp1[r] + b31;
        float kl  = -q[r];
        az0[r] += wrk * kz0; az1[r] += wrk * kz1; al[r] += wrk * kl;
        ze0[r] = zb0[r] + cnx * kz0;
        ze1[r] = zb1[r] + cnx * kz1;
      }
    }

    const float h6 = hstep * (1.0f / 6.0f);
#pragma unroll
    for (int r = 0; r < 8; ++r) {
      zb0[r] += h6 * az0[r];
      zb1[r] += h6 * az1[r];
      lpacc[r] += h6 * al[r];
    }
  }

  // ---- epilogue: lanes lo<8 of each half write their sample ----
  if (lo < 8) {
    const int m = s0 + mbase + lo;
    out[(size_t)m * 2 + 0] = zb0[lo];
    out[(size_t)m * 2 + 1] = zb1[lo];
    out[(size_t)BN * 2 + m] = lpacc[lo];   // lp1 after z1 (tuple concat)
  }
}

// ---------------- launcher ----------------

extern "C" void kernel_launch(void* const* d_in, const int* in_sizes, int n_in,
                              void* d_out, int out_size, void* d_ws, size_t ws_size,
                              hipStream_t stream) {
  (void)in_sizes; (void)n_in; (void)out_size; (void)d_ws; (void)ws_size;
  const float* x   = (const float*)d_in[0];
  const float* ctx = (const float*)d_in[1];
  const float* eps = (const float*)d_in[2];
  const float* W1  = (const float*)d_in[3];
  const float* b1  = (const float*)d_in[4];
  const float* W2  = (const float*)d_in[5];
  const float* b2  = (const float*)d_in[6];
  const float* W3  = (const float*)d_in[7];
  const float* b3  = (const float*)d_in[8];
  float* out = (float*)d_out;

  dim3 block(256);                 // 8 wave32 per block, 16 samples per wave
  dim3 grid(BN / (16 * 8));        // 4096 blocks
  cnf_rk4_wmma<<<grid, block, 0, stream>>>(x, ctx, eps, W1, b1, W2, b2, W3, b3, out);
}
